// MultiScaleGeometricAttention_18339510354285
// MI455X (gfx1250) — matrix operand order repaired and checked by reading, not compiled, not run on hardware
//
#include <hip/hip_runtime.h>

// ---------------------------------------------------------------------------
// MultiScaleGeometricAttention fused kernel for gfx1250 (MI455X, wave32, WMMA)
//   out = row_norm( exp(-cdist(x, positions)/eff_temp) ) @ values
// Two bf16 WMMA GEMMs fused flash-style over N=4096; P-chunk staging done by
// the Tensor Data Mover (tensor_load_to_lds, TENSORcnt) overlapped with GEMM2.
// ---------------------------------------------------------------------------

typedef __attribute__((ext_vector_type(16))) __bf16 v16bf;
typedef __attribute__((ext_vector_type(8)))  __bf16 v8bf;
typedef __attribute__((ext_vector_type(8)))  float  v8f;
typedef __attribute__((ext_vector_type(4)))  unsigned u32x4;
typedef __attribute__((ext_vector_type(8)))  int      i32x8;
typedef __attribute__((ext_vector_type(4)))  int      i32x4;

#define BT_TOT   16384          // B*T
#define DDIM     512
#define NPOS     4096
#define MTILE    64             // rows of x per workgroup
#define NCHUNK   64             // positions per chunk
#define NCHUNKS  (NPOS / NCHUNK)
#define XS_STRIDE 520           // halves; 1040B rows: 16B aligned, bank-shift 4
#define PS_STRIDE 520
#define WS_STRIDE 72            // halves; 144B rows: 16B aligned, bank-shift 36
#define ATT_SCALE 0.05f

union BF2  { __bf16 h[2]; unsigned u; };
union V16U { struct { v8bf lo, hi; } p; v16bf v; };

static __device__ __forceinline__ unsigned pack2(float a, float b) {
  BF2 t; t.h[0] = (__bf16)a; t.h[1] = (__bf16)b; return t.u;
}

static __device__ __forceinline__ v8f v8f_zero() {
  v8f r;
#pragma unroll
  for (int i = 0; i < 8; ++i) r[i] = 0.0f;
  return r;
}

// A-fragment (16x32 bf16): lane m holds 16 contiguous K = [k0+16h .. +16)
static __device__ __forceinline__ v16bf load_a_frag(const __bf16* p) {
  V16U u;
  u.p.lo = *(const v8bf*)(p);
  u.p.hi = *(const v8bf*)(p + 8);
  return u.v;
}

// B-fragment (32x16 bf16): lane n holds 16 contiguous K halves
static __device__ __forceinline__ v16bf load_b_frag(const __bf16* p) {
  V16U u;
  u.p.lo = *(const v8bf*)(p);
  u.p.hi = *(const v8bf*)(p + 8);
  return u.v;
}

static __device__ __forceinline__ v8f wmma_bf16(v16bf a, v16bf b, v8f c) {
  return __builtin_amdgcn_wmma_f32_16x16x32_bf16(false, a, false, b,
                                                 (short)0, c, false, false);
}

// ---------------------------------------------------------------------------
// TDM: DMA one 64-row x 512-half bf16 tile from global into LDS, inserting
// 4 DWORDs of padding every 256 DWORDs (-> 520-half LDS row stride).
// D# group0: count=1 | lds_addr | global_addr[56:0] | type=2
// D# group1: data_size=2B, pad_enable, pad_interval=7(256dw), pad_amount=3(4dw)
//            tensor_dim0=512, tensor_dim1=4096, tile_dim0=512, tile_dim1=64,
//            tensor_dim0_stride=512
// ---------------------------------------------------------------------------
static __device__ __forceinline__ void tdm_load_p_chunk(const __bf16* gsrc,
                                                        void* ldst) {
  unsigned long long ga = (unsigned long long)(uintptr_t)gsrc;
  unsigned lds = (unsigned)(uintptr_t)ldst;   // flat addr[31:0] == LDS offset
  u32x4 g0;
  g0[0] = 1u;                                  // count=1, user mode
  g0[1] = lds;                                 // lds_addr
  g0[2] = (unsigned)(ga & 0xffffffffu);        // global_addr[31:0]
  g0[3] = (unsigned)((ga >> 32) & 0x1ffffffu)  // global_addr[56:32]
          | (2u << 30);                        // type = 2 ("image")
  i32x8 g1;
  g1[0] = (1 << 16)                            // data_size = 2 bytes
        | (1 << 20)                            // pad_enable
        | (7 << 22)                            // pad_interval: 256 DWORDs
        | (3 << 25);                           // pad_amount:   4 DWORDs
  g1[1] = (512 << 16);                         // tensor_dim0[15:0] = 512
  g1[2] = (int)(4096u << 16);                  // tensor_dim1[15:0] = 4096
  g1[3] = (512 << 16);                         // tile_dim0 = 512
  g1[4] = 64;                                  // tile_dim1 = 64
  g1[5] = 512;                                 // tensor_dim0_stride = 512
  g1[6] = 0;
  g1[7] = 0;
  i32x4 gz = {0, 0, 0, 0};
  i32x8 gz8 = {0, 0, 0, 0, 0, 0, 0, 0};
  __builtin_amdgcn_tensor_load_to_lds(g0, g1, gz, gz, gz8, 0);
}

// ---------------------------------------------------------------------------
// Kernel A: precompute  p2[n], inv_eff[n], x2[m];  positions -> bf16 row-major
//           values -> bf16 TRANSPOSED  vbfT[D][N]  (makes GEMM2 B contiguous)
// ---------------------------------------------------------------------------
__global__ __launch_bounds__(256) void msga_prep(
    const float* __restrict__ x, const float* __restrict__ positions,
    const float* __restrict__ values, const float* __restrict__ temperature,
    float* __restrict__ p2f, float* __restrict__ ivef, float* __restrict__ x2f,
    __bf16* __restrict__ pbf, __bf16* __restrict__ vbfT)
{
  const int tid  = threadIdx.x;
  const int lane = tid & 31;
  const int wv   = tid >> 5;
  const int blk  = blockIdx.x;

  if (blk < 512) {                       // positions: bf16 copy + p2 + inv_eff
    int n = blk * 8 + wv;
    const float* src = positions + n * DDIM;
    float ss = 0.f;
#pragma unroll
    for (int j = 0; j < 4; ++j) {
      float4 v = ((const float4*)src)[j * 32 + lane];
      ss += v.x * v.x + v.y * v.y + v.z * v.z + v.w * v.w;
      uint2 pk; pk.x = pack2(v.x, v.y); pk.y = pack2(v.z, v.w);
      ((uint2*)(pbf + n * DDIM))[j * 32 + lane] = pk;
    }
#pragma unroll
    for (int m = 16; m >= 1; m >>= 1) ss += __shfl_xor(ss, m, 32);
    if (lane == 0) {
      p2f[n]  = ss;
      ivef[n] = 1.0f / ((fabsf(temperature[n]) + 0.1f) * ATT_SCALE);
    }
  } else if (blk < 2560) {               // x row norms
    int m = (blk - 512) * 8 + wv;
    const float* src = x + m * DDIM;
    float ss = 0.f;
#pragma unroll
    for (int j = 0; j < 4; ++j) {
      float4 v = ((const float4*)src)[j * 32 + lane];
      ss += v.x * v.x + v.y * v.y + v.z * v.z + v.w * v.w;
    }
#pragma unroll
    for (int mm = 16; mm >= 1; mm >>= 1) ss += __shfl_xor(ss, mm, 32);
    if (lane == 0) x2f[m] = ss;
  } else {                               // values: 64x64 tile transpose -> bf16
    __shared__ __attribute__((aligned(16))) __bf16 tile[64 * 68];
    int t  = blk - 2560;
    int n0 = (t & 63) * 64;
    int d0 = (t >> 6) * 64;
#pragma unroll
    for (int i = 0; i < 4; ++i) {
      int idx = i * 256 + tid;           // 0..1023 over 64 rows x 16 float4
      int r   = idx >> 4;
      int c4  = idx & 15;
      float4 v = ((const float4*)(values + (n0 + r) * DDIM + d0))[c4];
      uint2 pk; pk.x = pack2(v.x, v.y); pk.y = pack2(v.z, v.w);
      *(uint2*)&tile[r * 68 + c4 * 4] = pk;
    }
    __syncthreads();
#pragma unroll
    for (int i = 0; i < 4; ++i) {
      int idx = i * 256 + tid;
      int dr  = idx >> 4;                // output row (d)
      int nb  = (idx & 15) * 4;          // 4 halves along n
      BF2 a, b;
      a.h[0] = tile[(nb + 0) * 68 + dr];
      a.h[1] = tile[(nb + 1) * 68 + dr];
      b.h[0] = tile[(nb + 2) * 68 + dr];
      b.h[1] = tile[(nb + 3) * 68 + dr];
      uint2 pk; pk.x = a.u; pk.y = b.u;
      *(uint2*)(vbfT + (d0 + dr) * NPOS + n0 + nb) = pk;
    }
  }
}

// ---------------------------------------------------------------------------
// Kernel B: fused attention. 256 WGs x 256 threads (8 wave32).
//   Per WG: 64 rows of x resident in LDS (bf16). Loop over 64 chunks of N:
//   GEMM1 (WMMA bf16) -> exp weights -> LDS -> GEMM2 (WMMA bf16, persistent
//   64x64 f32 accumulators per wave, B streamed from L2-resident vbfT).
//   P chunks DMA'd into LDS by the Tensor Data Mover, overlapped with GEMM2.
// ---------------------------------------------------------------------------
__global__ __launch_bounds__(256) void msga_fused(
    const float* __restrict__ x, float* __restrict__ out,
    const __bf16* __restrict__ pbf, const __bf16* __restrict__ vbfT,
    const float* __restrict__ p2f, const float* __restrict__ ivef,
    const float* __restrict__ x2f)
{
  __shared__ __attribute__((aligned(16))) __bf16 Xs[MTILE * XS_STRIDE];
  __shared__ __attribute__((aligned(16))) __bf16 Ps[NCHUNK * PS_STRIDE];
  __shared__ __attribute__((aligned(16))) __bf16 Ws[MTILE * WS_STRIDE];
  __shared__ float rs2[MTILE][2];

  const int tid  = threadIdx.x;
  const int lane = tid & 31;
  const int ln   = lane & 15;
  const int hh   = lane >> 4;           // 0/1 lane half
  const int wv   = tid >> 5;            // wave 0..7
  const int m0   = blockIdx.x * MTILE;

  // ---- kick off TDM DMA of P chunk 0 (overlaps with X staging below) ----
  if (wv == 0) tdm_load_p_chunk(pbf, &Ps[0]);

  // ---- stage X tile: f32 global -> bf16 LDS (fully coalesced b128) ----
#pragma unroll
  for (int i = 0; i < 32; ++i) {
    int idx = i * 256 + tid;            // 8192 float4 total
    int r   = idx >> 7;                 // 128 float4 per row
    int c4  = idx & 127;
    float4 v = ((const float4*)(x + (m0 + r) * DDIM))[c4];
    uint2 pk; pk.x = pack2(v.x, v.y); pk.y = pack2(v.z, v.w);
    *(uint2*)&Xs[r * XS_STRIDE + c4 * 4] = pk;
  }

  const int mb1 = wv >> 1;              // GEMM1 row block (16 rows)
  const int nhb = (wv & 1) * 32;        // GEMM1 column half of the chunk

  float xr[8];                          // ||x||^2 for this wave's GEMM1 rows
#pragma unroll
  for (int j = 0; j < 8; ++j) xr[j] = x2f[m0 + 16 * mb1 + j + 8 * hh];

  float part[8];                        // per-thread row-sum partials
#pragma unroll
  for (int j = 0; j < 8; ++j) part[j] = 0.0f;

  v8f acc[16];                          // persistent O tile: 4 mb x 4 db
#pragma unroll
  for (int i = 0; i < 16; ++i) acc[i] = v8f_zero();

  const __bf16* vT_base = vbfT + (wv * 64) * NPOS;  // wave's D-slice

  if (wv == 0) __builtin_amdgcn_s_wait_tensorcnt(0);
  __syncthreads();                      // Xs and Ps(chunk 0) ready

  for (int c = 0; c < NCHUNKS; ++c) {
    // ---- GEMM1: S[16x32] = Xtile @ Pchunk^T over K=512 ----
    v8f S0 = v8f_zero(), S1 = v8f_zero();
#pragma unroll
    for (int ks = 0; ks < 16; ++ks) {
      int k0 = ks * 32;
      v16bf a  = load_a_frag(&Xs[(16 * mb1 + ln) * XS_STRIDE + k0 + 16 * hh]);
      v16bf b0 = load_b_frag(&Ps[(nhb + ln) * PS_STRIDE + k0 + 16 * hh]);
      v16bf b1 = load_b_frag(&Ps[(nhb + 16 + ln) * PS_STRIDE + k0 + 16 * hh]);
      S0 = wmma_bf16(a, b0, S0);
      S1 = wmma_bf16(a, b1, S1);
    }

    // ---- weights: w = exp(-sqrt(max(x2+p2-2S,0)) * inv_eff) ----
#pragma unroll
    for (int st = 0; st < 2; ++st) {
      int   nloc = nhb + 16 * st + ln;
      int   ng   = c * NCHUNK + nloc;
      float p2v  = p2f[ng];
      float iv   = ivef[ng];
#pragma unroll
      for (int j = 0; j < 8; ++j) {
        float s   = st ? S1[j] : S0[j];
        float sq  = fmaxf(xr[j] + p2v - 2.0f * s, 0.0f);
        float wgt = __expf(-sqrtf(sq) * iv);
        part[j]  += wgt;
        Ws[(16 * mb1 + j + 8 * hh) * WS_STRIDE + nloc] = (__bf16)wgt;
      }
    }
    __syncthreads();                    // Ws ready; Ps fully consumed

    // ---- TDM: DMA next P chunk into LDS, overlapped with GEMM2 ----
    if (c + 1 < NCHUNKS) {
      if (wv == 0)
        tdm_load_p_chunk(pbf + (c + 1) * NCHUNK * DDIM, &Ps[0]);
      // prefetch next chunk's vbfT lines for GEMM2's global B stream
      __builtin_prefetch(vbfT + (tid * 2) * NPOS + (c + 1) * NCHUNK, 0, 1);
      __builtin_prefetch(vbfT + (tid * 2 + 1) * NPOS + (c + 1) * NCHUNK, 0, 1);
    }

    // ---- GEMM2: O[64 x 64-slice] += W[64x64] @ V[64 x slice] ----
#pragma unroll
    for (int kk = 0; kk < 2; ++kk) {
      int k0 = kk * 32;
      v16bf aW[4], bV[4];
#pragma unroll
      for (int mb = 0; mb < 4; ++mb)
        aW[mb] = load_a_frag(&Ws[(16 * mb + ln) * WS_STRIDE + k0 + 16 * hh]);
#pragma unroll
      for (int db = 0; db < 4; ++db)
        bV[db] = load_b_frag(vT_base + (db * 16 + ln) * NPOS +
                             c * NCHUNK + k0 + 16 * hh);
#pragma unroll
      for (int mb = 0; mb < 4; ++mb)
#pragma unroll
        for (int db = 0; db < 4; ++db)
          acc[mb * 4 + db] = wmma_bf16(aW[mb], bV[db], acc[mb * 4 + db]);
    }

    if (c + 1 < NCHUNKS) {
      if (wv == 0) __builtin_amdgcn_s_wait_tensorcnt(0);
      __syncthreads();                  // Ps(c+1) visible; Ws consumed
    }
  }

  // ---- deterministic row-sum reduction (shuffle within 16-lane groups) ----
#pragma unroll
  for (int j = 0; j < 8; ++j) {
    float v = part[j];
    v += __shfl_xor(v, 1, 32);
    v += __shfl_xor(v, 2, 32);
    v += __shfl_xor(v, 4, 32);
    v += __shfl_xor(v, 8, 32);
    part[j] = v;
  }
  if (ln == 0) {
#pragma unroll
    for (int j = 0; j < 8; ++j)
      rs2[16 * mb1 + j + 8 * hh][wv & 1] = part[j];
  }
  __syncthreads();

  // ---- normalize and store output (coalesced 64B per lane-half) ----
#pragma unroll
  for (int mb = 0; mb < 4; ++mb) {
#pragma unroll
    for (int j = 0; j < 8; ++j) {
      int   m  = 16 * mb + j + 8 * hh;
      float sc = 1.0f / (rs2[m][0] + rs2[m][1] + 1e-8f);
#pragma unroll
      for (int db = 0; db < 4; ++db)
        out[(m0 + m) * DDIM + wv * 64 + db * 16 + ln] = acc[mb * 4 + db][j] * sc;
    }
  }
}

// ---------------------------------------------------------------------------
extern "C" void kernel_launch(void* const* d_in, const int* in_sizes, int n_in,
                              void* d_out, int out_size, void* d_ws, size_t ws_size,
                              hipStream_t stream) {
  const float* x           = (const float*)d_in[0];
  const float* positions   = (const float*)d_in[1];
  const float* values      = (const float*)d_in[2];
  const float* temperature = (const float*)d_in[3];
  float* out = (float*)d_out;

  // workspace layout: p2[4096] | inv_eff[4096] | x2[16384] | pbf | vbfT
  float*  p2f  = (float*)d_ws;
  float*  ivef = p2f + NPOS;
  float*  x2f  = ivef + NPOS;
  __bf16* pbf  = (__bf16*)(x2f + BT_TOT);
  __bf16* vbfT = pbf + (size_t)NPOS * DDIM;

  msga_prep<<<3072, 256, 0, stream>>>(x, positions, values, temperature,
                                      p2f, ivef, x2f, pbf, vbfT);
  msga_fused<<<BT_TOT / MTILE, 256, 0, stream>>>(x, out, pbf, vbfT,
                                                 p2f, ivef, x2f);
}
